// DGCNN_7129645711689
// MI455X (gfx1250) — compile-verified, matrix operand order checked by reading, not run on hardware
//
#include <hip/hip_runtime.h>

// ---------------------------------------------------------------------------
// DGCNN forward for MI455X (gfx1250, wave32, WMMA bf16 16x16x32)
// ---------------------------------------------------------------------------

typedef __bf16 bf16;
typedef bf16  v16bf __attribute__((ext_vector_type(16)));
typedef float v8f   __attribute__((ext_vector_type(8)));

#define BNI 0.99999500003749968f   // 1/sqrt(1+1e-5)
#define NPTS 2048
#define NB   8
#define KNN  20

union FragU { uint4 q[2]; v16bf v; };

// A/B fragment (16x32 / 32x16 bf16) from LDS row: row-major, row points at [row][kb].
// lane<16 -> khalf=0 covers K {0..7,16..23}; lane>=16 -> khalf=8 covers K {8..15,24..31}
__device__ inline v16bf load_frag_lds(const bf16* rowp, int khalf) {
  FragU f;
  f.q[0] = *(const uint4*)(rowp + khalf);
  f.q[1] = *(const uint4*)(rowp + khalf + 16);
  return f.v;
}

// B fragment column from global f32 (contiguous along K), converted to bf16.
__device__ inline v16bf load_frag_f32(const float* p, int khalf) {
  float4 f0 = *(const float4*)(p + khalf);
  float4 f1 = *(const float4*)(p + khalf + 4);
  float4 f2 = *(const float4*)(p + khalf + 16);
  float4 f3 = *(const float4*)(p + khalf + 20);
  v16bf v;
  v[0]=(bf16)f0.x; v[1]=(bf16)f0.y; v[2]=(bf16)f0.z; v[3]=(bf16)f0.w;
  v[4]=(bf16)f1.x; v[5]=(bf16)f1.y; v[6]=(bf16)f1.z; v[7]=(bf16)f1.w;
  v[8]=(bf16)f2.x; v[9]=(bf16)f2.y; v[10]=(bf16)f2.z; v[11]=(bf16)f2.w;
  v[12]=(bf16)f3.x; v[13]=(bf16)f3.y; v[14]=(bf16)f3.z; v[15]=(bf16)f3.w;
  return v;
}

__device__ inline void atomicMaxFloat(float* addr, float val) {
  int* ia = (int*)addr;
  int old = __float_as_int(*addr);
  while (__int_as_float(old) < val) {
    int assumed = old;
    old = atomicCAS(ia, assumed, __float_as_int(val));
    if (old == assumed) break;
  }
}

__device__ inline float leaky(float h) { return h >= 0.0f ? h : 0.2f * h; }

// ---------------------------------------------------------------------------
// (B,3,N) -> (B,N,3)
// ---------------------------------------------------------------------------
__global__ void transpose_x_kernel(const float* __restrict__ x, float* __restrict__ X0) {
  int i = blockIdx.x * 256 + threadIdx.x;
  if (i >= NB * NPTS * 3) return;
  int b = i / (NPTS * 3);
  int rem = i % (NPTS * 3);
  int n = rem / 3, c = rem % 3;
  X0[i] = x[((size_t)b * 3 + c) * NPTS + n];
}

// ---------------------------------------------------------------------------
// kNN: per (point, batch) block, distances in LDS, 20x argmax selection.
// pd = -||x_n - x_m||^2  == 2<x_n,x_m> - |x_n|^2 - |x_m|^2  (same ordering)
// ---------------------------------------------------------------------------
template <int CIN>
__global__ void __launch_bounds__(256) knn_kernel(const float* __restrict__ X,
                                                  int* __restrict__ idxout) {
  int n = blockIdx.x, b = blockIdx.y, tid = threadIdx.x;
  __shared__ float pd[NPTS];
  __shared__ float ctr[CIN];
  __shared__ float rv[256];
  __shared__ int   ri[256];
  for (int c = tid; c < CIN; c += 256) ctr[c] = X[((size_t)(b * NPTS) + n) * CIN + c];
  __syncthreads();
  for (int m = tid; m < NPTS; m += 256) {
    const float* xm = X + ((size_t)(b * NPTS) + m) * CIN;
    float d = 0.0f;
    for (int c = 0; c < CIN; ++c) { float df = xm[c] - ctr[c]; d += df * df; }
    pd[m] = -d;
  }
  __syncthreads();
  for (int it = 0; it < KNN; ++it) {
    float bvv = -3.4e38f; int bi = 0;
    for (int m = tid; m < NPTS; m += 256) {
      float v = pd[m];
      if (v > bvv) { bvv = v; bi = m; }
    }
    rv[tid] = bvv; ri[tid] = bi;
    __syncthreads();
    for (int s = 128; s > 0; s >>= 1) {
      if (tid < s) {
        if (rv[tid + s] > rv[tid] || (rv[tid + s] == rv[tid] && ri[tid + s] < ri[tid])) {
          rv[tid] = rv[tid + s]; ri[tid] = ri[tid + s];
        }
      }
      __syncthreads();
    }
    if (tid == 0) {
      int wsel = ri[0];
      idxout[((size_t)(b * NPTS) + n) * KNN + it] = wsel;
      pd[wsel] = -3.4e38f;
    }
    __syncthreads();
  }
}

// ---------------------------------------------------------------------------
// Fused EdgeConv: gather (nb-ctr | ctr) into LDS bf16, WMMA GEMM against w^T,
// BN + leaky, atomic max over the 20 neighbors, write (B,N,COUT).
// ---------------------------------------------------------------------------
template <int CIN, int COUT>
__global__ void __launch_bounds__(256) edgeconv_kernel(const float* __restrict__ X,
                                                       const int* __restrict__ idxb,
                                                       const float* __restrict__ w,
                                                       const float* __restrict__ g,
                                                       const float* __restrict__ bvec,
                                                       float* __restrict__ out) {
  constexpr int MT = 8;                 // points per block
  constexpr int EDGES = MT * KNN;       // 160 edge rows
  constexpr int MTILES = EDGES / 16;    // 10
  constexpr int KD = (2 * CIN < 32) ? 32 : 2 * CIN;
  constexpr bool SMALL = (2 * CIN < 32);
  constexpr int NCT = COUT / 16;
  static_assert(KD % 32 == 0, "K must be a multiple of 32");

  int b = blockIdx.y, tid = threadIdx.x;
  int n0 = blockIdx.x * MT;

  __shared__ int   ids[EDGES];
  __shared__ float maxbuf[MT * COUT];
  __shared__ bf16  featD[SMALL ? 1 : EDGES * CIN];   // neighbor - center half
  __shared__ float featCf[SMALL ? 1 : MT * CIN];     // center (f32, for diffs)
  __shared__ bf16  featCb[SMALL ? 1 : MT * CIN];     // center half (bf16, A-frag)
  __shared__ bf16  featS[SMALL ? EDGES * KD : 1];    // full padded feat (layer 1)
  __shared__ bf16  wpad[SMALL ? COUT * KD : 1];      // zero-padded weights (layer 1)

  for (int e = tid; e < EDGES; e += 256)
    ids[e] = idxb[((size_t)(b * NPTS) + n0 + e / KNN) * KNN + e % KNN];
  for (int e = tid; e < MT * COUT; e += 256) maxbuf[e] = -3.4e38f;
  if constexpr (SMALL) {
    for (int e = tid; e < COUT * KD; e += 256) {
      int c = e % KD;
      wpad[e] = (bf16)((c < 2 * CIN) ? w[(e / KD) * (2 * CIN) + c] : 0.0f);
    }
  } else {
    for (int e = tid; e < MT * CIN; e += 256) {
      float v = X[((size_t)(b * NPTS) + n0 + e / CIN) * CIN + e % CIN];
      featCf[e] = v; featCb[e] = (bf16)v;
    }
  }
  __syncthreads();
  if constexpr (SMALL) {
    for (int e = tid; e < EDGES * KD; e += 256) {
      int r = e / KD, c = e % KD, p = r / KNN;
      float v = 0.0f;
      if (c < CIN)
        v = X[((size_t)(b * NPTS) + ids[r]) * CIN + c] -
            X[((size_t)(b * NPTS) + n0 + p) * CIN + c];
      else if (c < 2 * CIN)
        v = X[((size_t)(b * NPTS) + n0 + p) * CIN + (c - CIN)];
      featS[e] = (bf16)v;
    }
  } else {
    for (int e = tid; e < EDGES * CIN; e += 256) {
      int r = e / CIN, c = e % CIN;
      float v = X[((size_t)(b * NPTS) + ids[r]) * CIN + c] - featCf[(r / KNN) * CIN + c];
      featD[e] = (bf16)v;
    }
  }
  __syncthreads();

  int lane = tid & 31, wave = tid >> 5;
  int khalf = (lane >> 4) * 8;
  int arow = lane & 15;
  for (int t = wave; t < MTILES * NCT; t += 8) {
    int mt = t / NCT, ot = t % NCT;
    int obase = ot * 16;
    int ocol = obase + (lane & 15);
    int edge = mt * 16 + arow;
    int pA = edge / KNN;
    v8f acc = {};
    for (int kb = 0; kb < KD; kb += 32) {
      v16bf a, bm;
      if constexpr (SMALL) {
        a  = load_frag_lds(featS + (size_t)edge * KD + kb, khalf);
        bm = load_frag_lds(wpad + (size_t)ocol * KD + kb, khalf);
      } else {
        const bf16* ap = (kb < CIN) ? (featD + (size_t)edge * CIN + kb)
                                    : (featCb + (size_t)pA * CIN + (kb - CIN));
        a  = load_frag_lds(ap, khalf);
        bm = load_frag_f32(w + (size_t)ocol * (2 * CIN) + kb, khalf);
      }
      acc = __builtin_amdgcn_wmma_f32_16x16x32_bf16(false, a, false, bm,
                                                    (short)0, acc, false, false);
    }
    float gg = g[ocol], bb = bvec[ocol];
    for (int e = 0; e < 8; ++e) {
      int m = e + ((lane >> 4) << 3);
      int pt = (mt * 16 + m) / KNN;
      float h = leaky(gg * acc[e] * BNI + bb);
      atomicMaxFloat(&maxbuf[pt * COUT + ocol], h);
    }
  }
  __syncthreads();
  for (int e = tid; e < MT * COUT; e += 256)
    out[((size_t)(b * NPTS) + n0 + e / COUT) * COUT + e % COUT] = maxbuf[e];
}

// ---------------------------------------------------------------------------
// h = leaky(gc * (Wc[1024x512] @ concat(x1..x4)) * BNI + bc); fused max/mean
// over N. One block per (b, 16-row o-tile); 8 waves split the 128 n-tiles.
// ---------------------------------------------------------------------------
__global__ void __launch_bounds__(256) conv_kernel(const float* __restrict__ x1,
                                                   const float* __restrict__ x2,
                                                   const float* __restrict__ x3,
                                                   const float* __restrict__ x4,
                                                   const float* __restrict__ wc,
                                                   const float* __restrict__ gc,
                                                   const float* __restrict__ bc,
                                                   float* __restrict__ x5,
                                                   float* __restrict__ x6) {
  int ot = blockIdx.x, b = blockIdx.y, tid = threadIdx.x;
  int lane = tid & 31, wave = tid >> 5;
  int obase = ot * 16;
  __shared__ bf16 a_lds[16 * 512];
  __shared__ float smax[8 * 16], ssum[8 * 16];
  for (int e = tid; e < 16 * 512; e += 256)
    a_lds[e] = (bf16)wc[(size_t)(obase + e / 512) * 512 + e % 512];
  __syncthreads();

  int khalf = (lane >> 4) * 8;
  float g8[8], b8[8], rmax[8], rsum[8];
  for (int e = 0; e < 8; ++e) {
    int o = obase + e + ((lane >> 4) << 3);
    g8[e] = gc[o]; b8[e] = bc[o];
    rmax[e] = -3.4e38f; rsum[e] = 0.0f;
  }
  const bf16* arow = a_lds + (size_t)(lane & 15) * 512;
  for (int nt = wave; nt < 128; nt += 8) {
    int n = nt * 16 + (lane & 15);
    v8f acc = {};
    for (int kb = 0; kb < 512; kb += 32) {
      v16bf a = load_frag_lds(arow + kb, khalf);
      const float* src; int Cb, cl;
      if (kb < 64)        { src = x1; Cb = 64;  cl = kb; }
      else if (kb < 128)  { src = x2; Cb = 64;  cl = kb - 64; }
      else if (kb < 256)  { src = x3; Cb = 128; cl = kb - 128; }
      else                { src = x4; Cb = 256; cl = kb - 256; }
      v16bf bm = load_frag_f32(src + ((size_t)(b * NPTS) + n) * Cb + cl, khalf);
      acc = __builtin_amdgcn_wmma_f32_16x16x32_bf16(false, a, false, bm,
                                                    (short)0, acc, false, false);
    }
    for (int e = 0; e < 8; ++e) {
      float h = leaky(g8[e] * acc[e] * BNI + b8[e]);
      rmax[e] = fmaxf(rmax[e], h);
      rsum[e] += h;
    }
  }
  for (int off = 8; off; off >>= 1)
    for (int e = 0; e < 8; ++e) {
      rmax[e] = fmaxf(rmax[e], __shfl_xor(rmax[e], off, 16));
      rsum[e] += __shfl_xor(rsum[e], off, 16);
    }
  if ((lane & 15) == 0) {
    int base = (lane >> 4) * 8;
    for (int e = 0; e < 8; ++e) {
      smax[wave * 16 + base + e] = rmax[e];
      ssum[wave * 16 + base + e] = rsum[e];
    }
  }
  __syncthreads();
  if (tid < 16) {
    float mx = -3.4e38f, sm = 0.0f;
    for (int w = 0; w < 8; ++w) { mx = fmaxf(mx, smax[w * 16 + tid]); sm += ssum[w * 16 + tid]; }
    x5[(size_t)b * 1024 + obase + tid] = mx;
    x6[(size_t)b * 1024 + obase + tid] = sm * (1.0f / (float)NPTS);
  }
}

// ---------------------------------------------------------------------------
// Tiny FC layers (one thread per output element; negligible FLOPs).
// ---------------------------------------------------------------------------
__global__ void fc1_kernel(const float* __restrict__ x5, const float* __restrict__ x6,
                           const float* __restrict__ wf1, const float* __restrict__ gf1,
                           const float* __restrict__ bf1, float* __restrict__ out) {
  int i = blockIdx.x * 256 + threadIdx.x;
  if (i >= NB * 512) return;
  int b = i / 512, o = i % 512;
  const float* wr = wf1 + (size_t)o * 2048;
  float s = 0.0f;
  for (int j = 0; j < 1024; ++j) s += x5[b * 1024 + j] * wr[j];
  for (int j = 0; j < 1024; ++j) s += x6[b * 1024 + j] * wr[1024 + j];
  out[i] = leaky(gf1[o] * s * BNI + bf1[o]);
}

__global__ void fc2_kernel(const float* __restrict__ h1, const float* __restrict__ wf2,
                           const float* __restrict__ bwf2, const float* __restrict__ gf2,
                           const float* __restrict__ bf2, float* __restrict__ out) {
  int i = blockIdx.x * 256 + threadIdx.x;
  if (i >= NB * 256) return;
  int b = i / 256, o = i % 256;
  const float* wr = wf2 + (size_t)o * 512;
  float s = 0.0f;
  for (int j = 0; j < 512; ++j) s += h1[b * 512 + j] * wr[j];
  out[i] = leaky(gf2[o] * (s + bwf2[o]) * BNI + bf2[o]);
}

__global__ void fc3_kernel(const float* __restrict__ h2, const float* __restrict__ wf3,
                           const float* __restrict__ bwf3, float* __restrict__ out) {
  int i = blockIdx.x * 256 + threadIdx.x;
  if (i >= NB * 40) return;
  int b = i / 40, o = i % 40;
  const float* wr = wf3 + (size_t)o * 256;
  float s = 0.0f;
  for (int j = 0; j < 256; ++j) s += h2[b * 256 + j] * wr[j];
  out[i] = s + bwf3[o];
}

// ---------------------------------------------------------------------------
extern "C" void kernel_launch(void* const* d_in, const int* in_sizes, int n_in,
                              void* d_out, int out_size, void* d_ws, size_t ws_size,
                              hipStream_t stream) {
  (void)in_sizes; (void)n_in; (void)out_size; (void)ws_size;
  const float* x    = (const float*)d_in[0];
  const float* w1   = (const float*)d_in[1];
  const float* g1   = (const float*)d_in[2];
  const float* b1   = (const float*)d_in[3];
  const float* w2   = (const float*)d_in[4];
  const float* g2   = (const float*)d_in[5];
  const float* b2   = (const float*)d_in[6];
  const float* w3   = (const float*)d_in[7];
  const float* g3   = (const float*)d_in[8];
  const float* b3   = (const float*)d_in[9];
  const float* w4   = (const float*)d_in[10];
  const float* g4   = (const float*)d_in[11];
  const float* b4   = (const float*)d_in[12];
  const float* wc   = (const float*)d_in[13];
  const float* gc   = (const float*)d_in[14];
  const float* bc   = (const float*)d_in[15];
  const float* wf1  = (const float*)d_in[16];
  const float* gf1  = (const float*)d_in[17];
  const float* bf1  = (const float*)d_in[18];
  const float* wf2  = (const float*)d_in[19];
  const float* bwf2 = (const float*)d_in[20];
  const float* gf2  = (const float*)d_in[21];
  const float* bf2  = (const float*)d_in[22];
  const float* wf3  = (const float*)d_in[23];
  const float* bwf3 = (const float*)d_in[24];
  float* out = (float*)d_out;

  float* ws = (float*)d_ws;
  float* X0 = ws;  ws += NB * NPTS * 3;
  float* X1 = ws;  ws += NB * NPTS * 64;
  float* X2 = ws;  ws += NB * NPTS * 64;
  float* X3 = ws;  ws += NB * NPTS * 128;
  float* X4 = ws;  ws += NB * NPTS * 256;
  int*  idx = (int*)ws; ws += NB * NPTS * KNN;
  float* x5 = ws;  ws += NB * 1024;
  float* x6 = ws;  ws += NB * 1024;
  float* h1 = ws;  ws += NB * 512;
  float* h2 = ws;  ws += NB * 256;

  dim3 blk(256);
  transpose_x_kernel<<<(NB * NPTS * 3 + 255) / 256, blk, 0, stream>>>(x, X0);

  knn_kernel<3><<<dim3(NPTS, NB), blk, 0, stream>>>(X0, idx);
  edgeconv_kernel<3, 64><<<dim3(NPTS / 8, NB), blk, 0, stream>>>(X0, idx, w1, g1, b1, X1);

  knn_kernel<64><<<dim3(NPTS, NB), blk, 0, stream>>>(X1, idx);
  edgeconv_kernel<64, 64><<<dim3(NPTS / 8, NB), blk, 0, stream>>>(X1, idx, w2, g2, b2, X2);

  knn_kernel<64><<<dim3(NPTS, NB), blk, 0, stream>>>(X2, idx);
  edgeconv_kernel<64, 128><<<dim3(NPTS / 8, NB), blk, 0, stream>>>(X2, idx, w3, g3, b3, X3);

  knn_kernel<128><<<dim3(NPTS, NB), blk, 0, stream>>>(X3, idx);
  edgeconv_kernel<128, 256><<<dim3(NPTS / 8, NB), blk, 0, stream>>>(X3, idx, w4, g4, b4, X4);

  conv_kernel<<<dim3(64, NB), blk, 0, stream>>>(X1, X2, X3, X4, wc, gc, bc, x5, x6);

  fc1_kernel<<<(NB * 512 + 255) / 256, blk, 0, stream>>>(x5, x6, wf1, gf1, bf1, h1);
  fc2_kernel<<<(NB * 256 + 255) / 256, blk, 0, stream>>>(h1, wf2, bwf2, gf2, bf2, h2);
  fc3_kernel<<<(NB * 40 + 255) / 256, blk, 0, stream>>>(h2, wf3, bwf3, out);
}
